// MambaVisionMixer_34265249088188
// MI455X (gfx1250) — matrix operand order repaired and checked by reading, not compiled, not run on hardware
//
#include <hip/hip_runtime.h>
#include <hip/hip_bf16.h>
#include <math.h>

// ---- problem dims (fixed by reference) ----
#define D_MODEL 768
#define D_STATE 128
#define D_CONV  4
#define D_INNER 1536          // 2 * D_MODEL
#define DT_RANK 48
#define BATCH   8
#define SEQ     1024
#define M_TOKENS (BATCH * SEQ)            // 8192
#define DTBC_N  (DT_RANK + 2 * D_STATE)   // 304

typedef __attribute__((ext_vector_type(16))) __bf16 bf16x16;
typedef __attribute__((ext_vector_type(8)))  __bf16 bf16x8;
typedef __attribute__((ext_vector_type(8)))  float  floatx8;
typedef __attribute__((ext_vector_type(4)))  float  floatx4;
typedef __attribute__((ext_vector_type(4)))  unsigned int uint32x4;
typedef __attribute__((ext_vector_type(8)))  int    int32x8;
typedef __attribute__((ext_vector_type(4)))  int    int32x4;

#define MODE_PLAIN    0
#define MODE_SPLIT    1
#define MODE_SOFTPLUS 2

#define LDS_STRIDE 40   // bf16 per LDS row: 80 B; matches TDM pad (16 DW row + 4 DW pad)

// f32 -> bf16 bulk converter (weights are reused by 128 M-tiles; convert once).
__global__ __launch_bounds__(256)
void cvt_f32_to_bf16(const float* __restrict__ src, __bf16* __restrict__ dst, int n)
{
    int i = (blockIdx.x * 256 + threadIdx.x) * 8;
    if (i + 8 <= n) {   // all weight sizes are multiples of 8
        floatx4 a = *(const floatx4*)(src + i);
        floatx4 b = *(const floatx4*)(src + i + 4);
        bf16x8 o;
        #pragma unroll
        for (int j = 0; j < 4; ++j) { o[j] = (__bf16)a[j]; o[4 + j] = (__bf16)b[j]; }
        *(bf16x8*)(dst + i) = o;
    }
}

// Issue one TDM load of a 64x32 bf16 weight tile into LDS (D# per §8 of
// CDNA5 ISA).  OOB rows/cols (N edge, K=48 tail) read as 0;
// pad_interval=3 (16 DWORDs = one 64B row) + pad_amount=3 (4 DWORDs) gives the
// 80B LDS row stride the WMMA fragment loads expect.
__device__ __forceinline__
void tdm_load_w_tile(const __bf16* W, int K, int N, int n0, int k0, unsigned ldsAddr)
{
    unsigned long long gaddr = (unsigned long long)(uintptr_t)(W + (size_t)n0 * K + k0);
    unsigned td0 = (unsigned)(K - k0);   // valid elems per row from tile start
    unsigned td1 = (unsigned)(N - n0);   // valid rows from tile start
    uint32x4 g0;
    g0[0] = 1u;                                   // count=1, user descriptor
    g0[1] = ldsAddr;                              // lds_addr
    g0[2] = (unsigned)(gaddr & 0xffffffffu);      // global_addr[31:0]
    g0[3] = (unsigned)((gaddr >> 32) & 0x1ffffffu) | (2u << 30);  // addr[56:32] | type=2
    int32x8 g1;
    g1[0] = (1 << 16)        // data_size = 1 -> 2 bytes (bf16)
          | (1 << 20)        // pad_enable
          | (3 << 22)        // pad_interval: 16 DWORDs
          | (3 << 25);       // pad_amount:   4 DWORDs
    g1[1] = (int)((td0 & 0xffffu) << 16);                 // tensor_dim0[15:0]
    g1[2] = (int)((td0 >> 16) | ((td1 & 0xffffu) << 16)); // dim0[31:16] | dim1[15:0]
    g1[3] = (int)((td1 >> 16) | (32u << 16));             // dim1[31:16] | tile_dim0=32
    g1[4] = 64;                                           // tile_dim1=64, tile_dim2=0
    g1[5] = (int)(unsigned)K;                             // tensor_dim0_stride[31:0]
    g1[6] = 0;
    g1[7] = 0;
    int32x4 gz4 = {0, 0, 0, 0};
    int32x8 gz8 = {0, 0, 0, 0, 0, 0, 0, 0};
    __builtin_amdgcn_tensor_load_to_lds(g0, g1, gz4, gz4, gz8, 0);
}

// C[M,N] = A[M,K] (f32, row-major, stride lda) @ W[N,K]^T (bf16), fp32 accum.
// 128 threads = 4 waves; block tile 64x64; each wave: 2x2 WMMA tiles -> 4
// v_wmma per k-step with fragment reuse.  Software-pipelined, double-buffered:
// TDM(kt+1) + A global loads(kt+1) are in flight while tile kt's WMMAs run.
template<int MODE>
__global__ __launch_bounds__(128)
void gemm_bf16_wmma(const float* __restrict__ A, int lda,
                    const __bf16* __restrict__ W,
                    float* __restrict__ C0, float* __restrict__ C1,
                    const float* __restrict__ bias,
                    int M, int N, int K, int ldc, int splitN)
{
    __shared__ __bf16 lA[2][64 * LDS_STRIDE];
    __shared__ __bf16 lB[2][64 * LDS_STRIDE];

    const int m0     = blockIdx.y * 64;
    const int n0     = blockIdx.x * 64;
    const int tid    = threadIdx.x;
    const int wave   = tid >> 5;
    const int lane   = tid & 31;
    const int laneHi = lane >> 4;        // 0: lanes 0-15, 1: lanes 16-31
    const int lane16 = lane & 15;
    const int wm     = (wave >> 1) * 32;
    const int wn     = (wave & 1) * 32;

    // A staging: thread -> (row 0..63, 16-float k-segment 0/16).  All K users
    // (768/1536/48) are multiples of 16 -> segments all-or-nothing valid.
    const int srowA = tid >> 1;
    const int scolA = (tid & 1) * 16;
    const float* aSeg = A + (size_t)(m0 + srowA) * lda + scolA;
    const int lAoff = srowA * LDS_STRIDE + scolA;

    floatx8 acc[2][2];
    #pragma unroll
    for (int i = 0; i < 2; ++i)
        #pragma unroll
        for (int j = 0; j < 2; ++j)
            acc[i][j] = (floatx8){0.f, 0.f, 0.f, 0.f, 0.f, 0.f, 0.f, 0.f};

    const unsigned ldsB0 = (unsigned)(uintptr_t)&lB[0][0];
    const unsigned ldsB1 = (unsigned)(uintptr_t)&lB[1][0];

    const int nk = (K + 31) >> 5;

    // ---- prologue: stage tile 0 into buffer 0
    {
        bf16x8 av0 = {}, av1 = {};
        if (scolA + 16 <= K) {
            floatx4 f0 = *(const floatx4*)(aSeg);
            floatx4 f1 = *(const floatx4*)(aSeg + 4);
            floatx4 f2 = *(const floatx4*)(aSeg + 8);
            floatx4 f3 = *(const floatx4*)(aSeg + 12);
            #pragma unroll
            for (int j = 0; j < 4; ++j) {
                av0[j] = (__bf16)f0[j]; av0[4 + j] = (__bf16)f1[j];
                av1[j] = (__bf16)f2[j]; av1[4 + j] = (__bf16)f3[j];
            }
        }
        *(bf16x8*)&lA[0][lAoff]     = av0;
        *(bf16x8*)&lA[0][lAoff + 8] = av1;
        if (tid < 32) tdm_load_w_tile(W, K, N, n0, 0, ldsB0);
    }

    for (int kt = 0; kt < nk; ++kt) {
        const int cur = kt & 1;
        const int nxt = cur ^ 1;
        const bool more = (kt + 1 < nk);

        // ---- kick off next tile: A global loads into regs + TDM into buffer nxt.
        // Buffer nxt was last READ in iteration kt-1; those reads completed before
        // that iteration's closing barrier, which precedes these writes.
        floatx4 f0 = {}, f1 = {}, f2 = {}, f3 = {};
        bool aValid = false;
        if (more) {
            const int k1 = (kt + 1) * 32;
            aValid = (k1 + scolA + 16 <= K);
            if (aValid) {
                f0 = *(const floatx4*)(aSeg + k1);
                f1 = *(const floatx4*)(aSeg + k1 + 4);
                f2 = *(const floatx4*)(aSeg + k1 + 8);
                f3 = *(const floatx4*)(aSeg + k1 + 12);
            }
            if (tid < 32) tdm_load_w_tile(W, K, N, n0, k1, nxt ? ldsB1 : ldsB0);
            if (kt + 2 < nk) __builtin_prefetch(aSeg + (kt + 2) * 32, 0, 1);
        }

        // TENSORcnt is in-order per wave: <=1 means tile kt's DMA is complete
        // while tile kt+1's stays in flight.
        if (tid < 32) {
            if (more) __builtin_amdgcn_s_wait_tensorcnt(1);
            else      __builtin_amdgcn_s_wait_tensorcnt(0);
        }
        __syncthreads();   // publish TDM(kt) + A stores(kt) to all waves

        // ---- fragments (CDNA5 §7.12.2) as contiguous 16B LDS loads
        bf16x16 afrag[2], bfrag[2];
        #pragma unroll
        for (int i = 0; i < 2; ++i) {
            const __bf16* aRow = &lA[cur][(wm + 16 * i + lane16) * LDS_STRIDE];
            bf16x8 lo = *(const bf16x8*)(aRow + laneHi * 8);
            bf16x8 hi = *(const bf16x8*)(aRow + 16 + laneHi * 8);
            #pragma unroll
            for (int j = 0; j < 8; ++j) { afrag[i][j] = lo[j]; afrag[i][8 + j] = hi[j]; }
        }
        #pragma unroll
        for (int i = 0; i < 2; ++i) {
            const __bf16* bRow = &lB[cur][(wn + 16 * i + lane16) * LDS_STRIDE];
            bf16x8 lo = *(const bf16x8*)(bRow + laneHi * 8);
            bf16x8 hi = *(const bf16x8*)(bRow + 16 + laneHi * 8);
            #pragma unroll
            for (int j = 0; j < 8; ++j) { bfrag[i][j] = lo[j]; bfrag[i][8 + j] = hi[j]; }
        }

        #pragma unroll
        for (int i = 0; i < 2; ++i)
            #pragma unroll
            for (int j = 0; j < 2; ++j)
                acc[i][j] = __builtin_amdgcn_wmma_f32_16x16x32_bf16(
                    false, afrag[i], false, bfrag[j], (short)0, acc[i][j], false, false);

        // ---- finish staging A(kt+1): cvt + ds_store into buffer nxt
        if (more) {
            bf16x8 av0 = {}, av1 = {};
            if (aValid) {
                #pragma unroll
                for (int j = 0; j < 4; ++j) {
                    av0[j] = (__bf16)f0[j]; av0[4 + j] = (__bf16)f1[j];
                    av1[j] = (__bf16)f2[j]; av1[4 + j] = (__bf16)f3[j];
                }
            }
            *(bf16x8*)&lA[nxt][lAoff]     = av0;
            *(bf16x8*)&lA[nxt][lAoff + 8] = av1;
        }
        __syncthreads();   // reads of buffer cur done -> it may be overwritten next iter
    }

    // ---- epilogue: C/D layout: VGPR r -> (M = r + 8*laneHi, N = lane16)
    #pragma unroll
    for (int i = 0; i < 2; ++i) {
        #pragma unroll
        for (int j = 0; j < 2; ++j) {
            int col = n0 + wn + 16 * j + lane16;
            if (col >= N) continue;
            #pragma unroll
            for (int r = 0; r < 8; ++r) {
                int row = m0 + wm + 16 * i + r + laneHi * 8;
                float v = acc[i][j][r];
                if (MODE == MODE_PLAIN) {
                    C0[(size_t)row * ldc + col] = v;
                } else if (MODE == MODE_SPLIT) {
                    if (col < splitN) C0[(size_t)row * splitN + col] = v;
                    else              C1[(size_t)row * splitN + (col - splitN)] = v;
                } else { // MODE_SOFTPLUS: dt = softplus(acc + dt_proj_b)
                    float t  = v + bias[col];
                    float sp = (t > 20.f) ? t : log1pf(__expf(t));
                    C0[(size_t)row * ldc + col] = sp;
                }
            }
        }
    }
}

// Causal depthwise conv1d (pad=3 left) + SiLU.  Memory bound; coalesced over d.
__global__ __launch_bounds__(256)
void conv_silu_kernel(const float* __restrict__ x, const float* __restrict__ cw,
                      const float* __restrict__ cb, float* __restrict__ xc)
{
    int idx = blockIdx.x * 256 + threadIdx.x;
    if (idx >= BATCH * SEQ * D_INNER) return;
    int d = idx % D_INNER;
    int l = (idx / D_INNER) % SEQ;
    int b = idx / (D_INNER * SEQ);
    float acc = cb[d];
    #pragma unroll
    for (int k = 0; k < D_CONV; ++k) {
        int ls = l - (D_CONV - 1) + k;
        if (ls >= 0)
            acc += x[((size_t)(b * SEQ + ls)) * D_INNER + d] * cw[d * D_CONV + k];
    }
    xc[idx] = acc / (1.f + __expf(-acc));   // silu
}

// Selective scan: one wave32 per (batch, channel d); lane owns 4 of 128 states.
__global__ __launch_bounds__(256)
void selective_scan_kernel(const float* __restrict__ u,    const float* __restrict__ dt,
                           const float* __restrict__ dtBC, const float* __restrict__ A_log,
                           const float* __restrict__ Dp,   const float* __restrict__ res,
                           float* __restrict__ yg)
{
    int wid  = blockIdx.x * 8 + (threadIdx.x >> 5);   // 8 waves / block
    int lane = threadIdx.x & 31;
    int b    = wid / D_INNER;
    int d    = wid % D_INNER;
    int n0   = lane * 4;

    float a[4], s[4];
    #pragma unroll
    for (int j = 0; j < 4; ++j) {
        a[j] = -__expf(A_log[(size_t)d * D_STATE + n0 + j]);  // A = -exp(A_log)
        s[j] = 0.f;
    }
    float Dd = Dp[d];

    for (int l = 0; l < SEQ; ++l) {
        size_t tok = (size_t)b * SEQ + l;
        float dtv = dt[tok * D_INNER + d];
        float uv  = u [tok * D_INNER + d];
        const float* bc = dtBC + tok * DTBC_N;
        float part = 0.f;
        #pragma unroll
        for (int j = 0; j < 4; ++j) {
            float Bv = bc[DT_RANK + n0 + j];
            float Cv = bc[DT_RANK + D_STATE + n0 + j];
            s[j] = s[j] * __expf(dtv * a[j]) + dtv * uv * Bv;
            part += s[j] * Cv;
        }
        #pragma unroll
        for (int off = 16; off > 0; off >>= 1)
            part += __shfl_xor(part, off, 32);
        if (lane == 0) {
            float y = part + uv * Dd;
            float r = res[tok * D_INNER + d];          // read BEFORE aliased write
            yg[tok * D_INNER + d] = y * (r / (1.f + __expf(-r)));
        }
    }
}

extern "C" void kernel_launch(void* const* d_in, const int* in_sizes, int n_in,
                              void* d_out, int out_size, void* d_ws, size_t ws_size,
                              hipStream_t stream)
{
    const float* hidden     = (const float*)d_in[0];
    const float* in_proj_w  = (const float*)d_in[1];
    const float* conv_w     = (const float*)d_in[2];
    const float* conv_b     = (const float*)d_in[3];
    const float* x_proj_w   = (const float*)d_in[4];
    const float* dt_proj_w  = (const float*)d_in[5];
    const float* dt_proj_b  = (const float*)d_in[6];
    const float* A_log      = (const float*)d_in[7];
    const float* D_param    = (const float*)d_in[8];
    const float* out_proj_w = (const float*)d_in[9];
    float* out = (float*)d_out;

    // ---- workspace layout (floats)
    float* ws = (float*)d_ws;
    const size_t BLD = (size_t)BATCH * SEQ * D_INNER;   // 12,582,912
    float* xb   = ws;             // x (pre-conv); reused as dt afterwards
    float* resb = ws + BLD;       // res gate; yg aliases it (read-then-write, same thread)
    float* xcb  = ws + 2 * BLD;   // silu(conv(x)) == u
    float* dtBC = ws + 3 * BLD;   // [8192, 304]
    float* dtb  = xb;
    float* ygb  = resb;
    // bf16 weight copies (converted once, reused by all 128 M-tiles)
    const int NIN = (2 * D_INNER) * D_MODEL;   // 2,359,296
    const int NXP = DTBC_N * D_INNER;          //   466,944
    const int NDT = D_INNER * DT_RANK;         //    73,728
    const int NOUT = D_MODEL * D_INNER;        // 1,179,648
    __bf16* w_in  = (__bf16*)(ws + 3 * BLD + (size_t)M_TOKENS * DTBC_N);
    __bf16* w_x   = w_in + NIN;
    __bf16* w_dt  = w_x + NXP;
    __bf16* w_out = w_dt + NDT;

    // 0) weight conversion passes
    cvt_f32_to_bf16<<<(NIN / 8 + 255) / 256, 256, 0, stream>>>(in_proj_w, w_in, NIN);
    cvt_f32_to_bf16<<<(NXP / 8 + 255) / 256, 256, 0, stream>>>(x_proj_w, w_x, NXP);
    cvt_f32_to_bf16<<<(NDT / 8 + 255) / 256, 256, 0, stream>>>(dt_proj_w, w_dt, NDT);
    cvt_f32_to_bf16<<<(NOUT / 8 + 255) / 256, 256, 0, stream>>>(out_proj_w, w_out, NOUT);

    dim3 blk(128);

    // 1) in_proj: [8192,768] x [3072,768]^T -> split into x | res
    gemm_bf16_wmma<MODE_SPLIT><<<dim3((2 * D_INNER) / 64, M_TOKENS / 64), blk, 0, stream>>>(
        hidden, D_MODEL, w_in, xb, resb, nullptr,
        M_TOKENS, 2 * D_INNER, D_MODEL, 0, D_INNER);

    // 2) causal depthwise conv + SiLU
    conv_silu_kernel<<<(BATCH * SEQ * D_INNER + 255) / 256, 256, 0, stream>>>(
        xb, conv_w, conv_b, xcb);

    // 3) x_proj: [8192,1536] x [304,1536]^T -> dtBC (N edge handled by TDM OOB-zero)
    gemm_bf16_wmma<MODE_PLAIN><<<dim3((DTBC_N + 63) / 64, M_TOKENS / 64), blk, 0, stream>>>(
        xcb, D_INNER, w_x, dtBC, nullptr, nullptr,
        M_TOKENS, DTBC_N, D_INNER, DTBC_N, 0);

    // 4) dt_proj + softplus: A = dtBC[:, :48] (lda=304), K=48 tail zero-padded
    gemm_bf16_wmma<MODE_SOFTPLUS><<<dim3(D_INNER / 64, M_TOKENS / 64), blk, 0, stream>>>(
        dtBC, DTBC_N, w_dt, dtb, nullptr, dt_proj_b,
        M_TOKENS, D_INNER, DT_RANK, D_INNER, 0);

    // 5) selective scan + skip + SiLU(res) gating; 12288 waves = BATCH * D_INNER
    selective_scan_kernel<<<(BATCH * D_INNER) / 8, 256, 0, stream>>>(
        xcb, dtb, dtBC, A_log, D_param, resb, ygb);

    // 6) out_proj: [8192,1536] x [768,1536]^T -> d_out
    gemm_bf16_wmma<MODE_PLAIN><<<dim3(D_MODEL / 64, M_TOKENS / 64), blk, 0, stream>>>(
        ygb, D_INNER, w_out, out, nullptr, nullptr,
        M_TOKENS, D_MODEL, D_INNER, D_MODEL, 0);
}
